// SaliencyGuidedLoss_82678120448346
// MI455X (gfx1250) — compile-verified, matrix operand order checked by reading, not run on hardware
//
#include <hip/hip_runtime.h>
#include <hip/hip_bf16.h>
#include <math.h>

typedef __attribute__((ext_vector_type(16))) _Float16 v16h;
typedef __attribute__((ext_vector_type(8)))  float    v8f;

#define EPSF 1e-6f
#define NB 64
#define CCH 512
#define HW 784            // 28*28
#define IMH 224
#define IMW 224
#define NPIX_TOTAL (NB * HW)          // 50176
#define LOSS_BLOCKS ((NPIX_TOTAL + 255) / 256)  // 196

// ---------------- ws layout (floats) ----------------
// [0, 50176)                : attn_raw  [64][784]
// [50176, 50176+3211264)    : var map   [64][224][224]
// then 256 ints             : mm[0..63]=attn_min, [64..127]=attn_max,
//                             [128..191]=sal_min, [192..255]=sal_max (float bits)
// then 196 floats           : loss partials

__global__ __launch_bounds__(64) void k0_init(int* __restrict__ mm) {
    int t = threadIdx.x;
    if (t < 64) {
        mm[t]        = 0x7f800000;  // attn_min = +inf
        mm[64 + t]   = 0;           // attn_max = 0 (values >= 0)
        mm[128 + t]  = 0x7f800000;  // sal_min
        mm[192 + t]  = 0;           // sal_max
    }
}

// ---- K2: attn = sqrt(mean_c f^2) via WMMA (ones^T x F^2), one wave per 16 pixels ----
__global__ __launch_bounds__(32) void k2_attn(const float* __restrict__ feat,
                                              float* __restrict__ attn,
                                              int* __restrict__ mm) {
    const int wave = blockIdx.x;        // 64 * 49 waves
    const int b    = wave / 49;
    const int g    = wave % 49;         // 16-pixel group
    const int lane = threadIdx.x;
    const int col  = lane & 15;         // pixel within group (matrix column N)
    const int krow = (lane >> 4) * 16;  // lanes 16-31 hold K=16..31 of each 32-chunk

    const float* base = feat + (size_t)b * CCH * HW + (size_t)g * 16 + col;

    v16h a;
    #pragma unroll
    for (int j = 0; j < 16; ++j) a[j] = (_Float16)1.0f;
    v8f acc = {};

    for (int cb = 0; cb < CCH; cb += 32) {
        const float* p = base + (size_t)(cb + krow) * HW;
        if (cb + 32 < CCH)
            __builtin_prefetch((const void*)(p + (size_t)32 * HW), 0, 0);
        v16h bm;
        #pragma unroll
        for (int j = 0; j < 16; ++j) {
            float v = p[(size_t)j * HW];
            bm[j] = (_Float16)(v * v);
        }
        // D = ones(16x32) * F2(32x16) + C : every D row = per-column channel sums
        acc = __builtin_amdgcn_wmma_f32_16x16x32_f16(false, a, false, bm,
                                                     (short)0, acc, false, false);
    }

    float s      = acc[0];                       // lanes 0-15: (M=0,N=lane); 16-31 duplicate
    float attn_v = sqrtf(s * (1.0f / (float)CCH));
    if (lane < 16) attn[(size_t)b * HW + g * 16 + lane] = attn_v;

    float mn = attn_v, mx = attn_v;
    #pragma unroll
    for (int m = 16; m >= 1; m >>= 1) {
        mn = fminf(mn, __shfl_xor(mn, m, 32));
        mx = fmaxf(mx, __shfl_xor(mx, m, 32));
    }
    if (lane == 0) {
        atomicMin(&mm[b],      __float_as_int(mn));
        atomicMax(&mm[64 + b], __float_as_int(mx));
    }
}

// ---- K3: async-copy RGB tile+halo to LDS, gray -> 5x5 box variance (zero-pad SAME) ----
__global__ __launch_bounds__(256) void k3_var(const float* __restrict__ img,
                                              float* __restrict__ varmap,
                                              int* __restrict__ mm) {
    __shared__ float chR[36 * 36];
    __shared__ float chG[36 * 36];
    __shared__ float chB[36 * 36];
    __shared__ float gray[36 * 36];
    __shared__ float rmn[256];
    __shared__ float rmx[256];

    const int blk = blockIdx.x;         // 64 * 49
    const int b   = blk / 49;
    const int t   = blk % 49;
    const int y0  = (t / 7) * 32;
    const int x0  = (t % 7) * 32;
    const int tid = threadIdx.x;

    const float* ib = img + (size_t)b * 3 * IMH * IMW;
    const unsigned long long gbase = (unsigned long long)(uintptr_t)ib;

    // Stage 1: async global->LDS copies of the three channel tiles (GVS form:
    // uniform 64-bit base in SGPRs + per-lane 32-bit byte offset). Channel
    // selection goes into vaddr, NOT inst offset (inst offset shifts LDS too).
    for (int i = tid; i < 36 * 36; i += 256) {
        int ly = i / 36, lx = i % 36;
        int gy = y0 + ly - 2, gx = x0 + lx - 2;
        if (gy >= 0 && gy < IMH && gx >= 0 && gx < IMW) {
            unsigned voff = (unsigned)((gy * IMW + gx) * 4);
            unsigned l0 = (unsigned)(uintptr_t)&chR[i];   // LDS byte offset (addr[31:0])
            unsigned l1 = (unsigned)(uintptr_t)&chG[i];
            unsigned l2 = (unsigned)(uintptr_t)&chB[i];
            asm volatile("global_load_async_to_lds_b32 %0, %1, %2"
                         :: "v"(l0), "v"(voff), "s"(gbase) : "memory");
            asm volatile("global_load_async_to_lds_b32 %0, %1, %2"
                         :: "v"(l1), "v"(voff + 200704u), "s"(gbase) : "memory");
            asm volatile("global_load_async_to_lds_b32 %0, %1, %2"
                         :: "v"(l2), "v"(voff + 401408u), "s"(gbase) : "memory");
        }
    }
    // Wait for this wave's async transfers, then zero the out-of-image halo
    // (DS stores are unordered vs ASYNC writes, so zero only after the wait).
    asm volatile("s_wait_asynccnt 0x0" ::: "memory");
    for (int i = tid; i < 36 * 36; i += 256) {
        int ly = i / 36, lx = i % 36;
        int gy = y0 + ly - 2, gx = x0 + lx - 2;
        if (!(gy >= 0 && gy < IMH && gx >= 0 && gx < IMW)) {
            chR[i] = 0.0f; chG[i] = 0.0f; chB[i] = 0.0f;
        }
    }
    __syncthreads();   // all waves' asyncs waited before their barrier arrival

    for (int i = tid; i < 36 * 36; i += 256)
        gray[i] = 0.299f * chR[i] + 0.587f * chG[i] + 0.114f * chB[i];
    __syncthreads();

    float lmn = 3.402823466e+38f, lmx = 0.0f;
    #pragma unroll
    for (int q = 0; q < 4; ++q) {
        int pix = tid + q * 256;        // 0..1023
        int py = pix >> 5, px = pix & 31;
        float s1 = 0.0f, s2 = 0.0f;
        #pragma unroll
        for (int dy = 0; dy < 5; ++dy) {
            const float* row = &gray[(py + dy) * 36 + px];
            #pragma unroll
            for (int dx = 0; dx < 5; ++dx) {
                float g = row[dx];
                s1 += g;
                s2 += g * g;
            }
        }
        float lm  = s1 * (1.0f / 25.0f);
        float var = fmaxf(s2 * (1.0f / 25.0f) - lm * lm, 0.0f);
        varmap[((size_t)b * IMH + (y0 + py)) * IMW + (x0 + px)] = var;
        lmn = fminf(lmn, var);
        lmx = fmaxf(lmx, var);
    }

    rmn[tid] = lmn; rmx[tid] = lmx;
    __syncthreads();
    for (int s = 128; s > 0; s >>= 1) {
        if (tid < s) {
            rmn[tid] = fminf(rmn[tid], rmn[tid + s]);
            rmx[tid] = fmaxf(rmx[tid], rmx[tid + s]);
        }
        __syncthreads();
    }
    if (tid == 0) {
        atomicMin(&mm[128 + b], __float_as_int(rmn[0]));
        atomicMax(&mm[192 + b], __float_as_int(rmx[0]));
    }
}

// ---- K4: triangle(8x) resize of raw var, normalize both maps, (attn-sal)^2 partials ----
__global__ __launch_bounds__(256) void k4_loss(const float* __restrict__ attn,
                                               const float* __restrict__ varmap,
                                               const int* __restrict__ mm,
                                               float* __restrict__ partials) {
    __shared__ float red[256];
    const int idx = blockIdx.x * 256 + threadIdx.x;
    float acc = 0.0f;
    if (idx < NPIX_TOTAL) {
        const int b  = idx / HW;
        const int r  = idx % HW;
        const int oy = r / 28, ox = r % 28;

        // attn normalization
        float amn = __int_as_float(mm[b]);
        float amx = __int_as_float(mm[64 + b]);
        float ad  = amx - amn;
        float av  = attn[idx];
        float an  = (ad > EPSF) ? (av - amn) / ad : 0.5f;

        // antialiased bilinear 224->28: triangle kernel radius 8, 16 taps/axis,
        // edge weights renormalized (matches jax.image.resize weight matrices)
        float wy[16], wx[16];
        float sy = 0.0f, sx = 0.0f;
        #pragma unroll
        for (int k = 0; k < 16; ++k) {
            float w  = 1.0f - fabsf((float)k - 7.5f) * 0.125f;
            int iy = 8 * oy - 4 + k;
            int ix = 8 * ox - 4 + k;
            wy[k] = (iy >= 0 && iy < IMH) ? w : 0.0f; sy += wy[k];
            wx[k] = (ix >= 0 && ix < IMW) ? w : 0.0f; sx += wx[k];
        }
        const float* vb = varmap + (size_t)b * IMH * IMW;
        float raw = 0.0f;
        for (int k = 0; k < 16; ++k) {
            if (wy[k] == 0.0f) continue;
            int iy = 8 * oy - 4 + k;
            const float* rp = vb + (size_t)iy * IMW;
            float rowacc = 0.0f;
            #pragma unroll
            for (int u = 0; u < 16; ++u) {
                int ix = 8 * ox - 4 + u;
                int ic = ix < 0 ? 0 : (ix > IMW - 1 ? IMW - 1 : ix);  // wx[u]==0 when OOB
                rowacc += wx[u] * rp[ic];
            }
            raw += wy[k] * rowacc;
        }
        raw /= (sy * sx);

        float smn = __int_as_float(mm[128 + b]);
        float smx = __int_as_float(mm[192 + b]);
        float sd  = smx - smn;
        float sn  = (sd > EPSF) ? (raw - smn) / sd : 0.5f;

        float d = an - sn;
        acc = d * d;
    }
    red[threadIdx.x] = acc;
    __syncthreads();
    for (int s = 128; s > 0; s >>= 1) {
        if (threadIdx.x < s) red[threadIdx.x] += red[threadIdx.x + s];
        __syncthreads();
    }
    if (threadIdx.x == 0) partials[blockIdx.x] = red[0];
}

__global__ __launch_bounds__(256) void k5_final(const float* __restrict__ partials,
                                                int n, float* __restrict__ out) {
    __shared__ float red[256];
    float a = (threadIdx.x < n) ? partials[threadIdx.x] : 0.0f;
    red[threadIdx.x] = a;
    __syncthreads();
    for (int s = 128; s > 0; s >>= 1) {
        if (threadIdx.x < s) red[threadIdx.x] += red[threadIdx.x + s];
        __syncthreads();
    }
    if (threadIdx.x == 0) out[0] = red[0] * (0.15f / (float)NPIX_TOTAL);
}

extern "C" void kernel_launch(void* const* d_in, const int* in_sizes, int n_in,
                              void* d_out, int out_size, void* d_ws, size_t ws_size,
                              hipStream_t stream) {
    const float* feat = (const float*)d_in[0];   // [64,512,28,28] f32
    const float* img  = (const float*)d_in[1];   // [64,3,224,224] f32
    float* out        = (float*)d_out;

    float* ws_attn = (float*)d_ws;                                   // 50176
    float* ws_var  = ws_attn + NPIX_TOTAL;                           // 64*224*224
    int*   ws_mm   = (int*)(ws_var + (size_t)NB * IMH * IMW);        // 256 ints
    float* ws_part = (float*)(ws_mm + 256);                          // 196

    k0_init<<<1, 64, 0, stream>>>(ws_mm);
    k2_attn<<<NB * 49, 32, 0, stream>>>(feat, ws_attn, ws_mm);
    k3_var<<<NB * 49, 256, 0, stream>>>(img, ws_var, ws_mm);
    k4_loss<<<LOSS_BLOCKS, 256, 0, stream>>>(ws_attn, ws_var, ws_mm, ws_part);
    k5_final<<<1, 256, 0, stream>>>(ws_part, LOSS_BLOCKS, out);
}
